// LSTMCell_11905649345053
// MI455X (gfx1250) — matrix operand order repaired and checked by reading, not compiled, not run on hardware
//
#include <hip/hip_runtime.h>
#include <hip/hip_bf16.h>

typedef __attribute__((ext_vector_type(16))) __bf16 v16bf;
typedef __attribute__((ext_vector_type(8)))  float  v8f;

#define SEQ   512
#define BATCH 64
#define ISZ   512
#define HSZ   512
#define G4    2048           // 4*HSZ gate rows
#define KC    1024           // combined K = ISZ + HSZ
#define KT    32             // KC / 32 k-tiles
#define KSLICE 4             // K split across 4 waves per tile
#define KT_W  (KT / KSLICE)  // 8 k-tiles per wave
#define TILE_ELEMS 512       // 32x16 bf16 elements per packed B tile

// ---------------------------------------------------------------------------
// ws layout (bytes):
//   [0,        4 MB)         Wpack  bf16 (G4 x KC) in B-fragment tile order
//   [4 MB,     36 MB)        x_bf16 (SEQ*BATCH*ISZ)
//   [36 MB,    +64 KB)       h_bf16 buffer 0 (BATCH x HSZ)
//   [+64 KB,   +128 KB)      h_bf16 buffer 1
//   [+128 KB,  +256 KB)      c fp32 (BATCH x HSZ)
// ---------------------------------------------------------------------------

// Pack [W_ih | W_hh] (fp32, row-major (n,k)) into bf16 tiles laid out exactly
// as lstm_step's B-fragment loads read them (contiguous K, same order as A): tile (nt,kt) holds a 32x16
// (KxN) block; lane = hi*16 + (n&15) owns 16 contiguous K values.
__global__ void pack_weights(const float* __restrict__ w_ih,
                             const float* __restrict__ w_hh,
                             __bf16* __restrict__ wpack) {
  int idx = blockIdx.x * blockDim.x + threadIdx.x;   // [0, G4*KC)
  int n = idx >> 10;          // gate row
  int k = idx & (KC - 1);     // combined K
  float v = (k < ISZ) ? w_ih[(size_t)n * ISZ + k]
                      : w_hh[(size_t)n * HSZ + (k - ISZ)];
  int nt = n >> 4, nl = n & 15;
  int kt = k >> 5, kr = k & 31;
  int hi = kr >> 4, j = kr & 15;
  size_t o = (size_t)(nt * KT + kt) * TILE_ELEMS + (hi * 16 + nl) * 16 + j;
  wpack[o] = (__bf16)v;
}

__global__ void convert_input(const float* __restrict__ x,
                              __bf16* __restrict__ xb) {
  size_t idx = (size_t)blockIdx.x * blockDim.x + threadIdx.x;
  xb[idx] = (__bf16)x[idx];
}

__global__ void init_state(const float* __restrict__ h0,
                           const float* __restrict__ c0,
                           __bf16* __restrict__ hb,
                           float* __restrict__ c) {
  int idx = blockIdx.x * blockDim.x + threadIdx.x;   // [0, BATCH*HSZ)
  hb[idx] = (__bf16)h0[idx];
  c[idx]  = c0[idx];
}

// One LSTM time step. Grid = 64 blocks x 256 threads = 512 waves.
// Each (16 batch x 16 hidden) tile is owned by a group of 4 waves that split
// K=1024 into 4 slices of 256 (8 WMMA k-steps each) -> 4x shorter per-step
// critical path and minimal (non-redundant) L2 weight traffic. Each wave
// accumulates all four gates for its tile so i/f/g/o for a given (b,n) meet
// in one lane. Partial sums are combined through LDS; every wave then applies
// the nonlinearities for 2 of the 8 fragment rows.
__global__ void __launch_bounds__(256)
lstm_step(const __bf16* __restrict__ wpack,
          const __bf16* __restrict__ xb,      // x_bf16 + t*BATCH*ISZ
          const __bf16* __restrict__ hin,     // bf16 h_{t-1}
          __bf16* __restrict__ hout,          // bf16 h_t (other buffer)
          float* __restrict__ c,              // fp32 cell state (in/out)
          const float* __restrict__ bias_ih,
          const float* __restrict__ bias_hh,
          float* __restrict__ out_t,          // out + t*BATCH*HSZ
          float* __restrict__ hT,
          float* __restrict__ cT,
          int last) {
  __shared__ float redbuf[8 * 4 * 32 * 8];    // [wave][gate][lane][r] = 32 KB

  const int lane = threadIdx.x & 31;
  const int wv   = threadIdx.x >> 5;          // 0..7
  const int grp  = wv >> 2;                   // tile within block: 0..1
  const int ks   = wv & 3;                    // K slice: 0..3
  const int tile = blockIdx.x * 2 + grp;      // 0..127
  const int mt   = tile & 3;                  // batch tile
  const int ht   = tile >> 2;                 // hidden tile 0..31
  const int nl   = lane & 15;
  const int hi   = lane >> 4;
  const int arow = mt * 16 + nl;              // A-fragment batch row

  // slice 0 starts from the bias; other slices start from zero
  v8f acc[4];
#pragma unroll
  for (int g = 0; g < 4; ++g) {
    float b = 0.0f;
    if (ks == 0) {                             // wave-uniform condition
      int n = g * HSZ + ht * 16 + nl;
      b = bias_ih[n] + bias_hh[n];
    }
#pragma unroll
    for (int e = 0; e < 8; ++e) acc[g][e] = b;
  }

  // K range for this slice: [ks*256, ks*256+256). Slices 0-1 read x, 2-3 read h.
  const int kk0 = ks * KT_W;
  const __bf16* arow_src = (ks < 2) ? (xb  + (size_t)arow * ISZ + ks * 256)
                                    : (hin + (size_t)arow * HSZ + (ks - 2) * 256);
  int tbase[4];
#pragma unroll
  for (int g = 0; g < 4; ++g) tbase[g] = (g * 32 + ht) * KT + kk0;  // nt*KT + kk0

#pragma unroll
  for (int kk = 0; kk < KT_W; ++kk) {
    const __bf16* rp = arow_src + kk * 32;
    // A fragment (16x32 bf16): lane hi-half takes 16 contiguous K at offset hi*16
    v16bf a;
    ((uint4*)&a)[0] = *(const uint4*)(rp + hi * 16);
    ((uint4*)&a)[1] = *(const uint4*)(rp + hi * 16 + 8);
#pragma unroll
    for (int g = 0; g < 4; ++g) {
      const __bf16* bp = wpack + (size_t)(tbase[g] + kk) * TILE_ELEMS + lane * 16;
      if (kk + 1 < KT_W)
        __builtin_prefetch(wpack + (size_t)(tbase[g] + kk + 1) * TILE_ELEMS + lane * 16, 0, 1);
      v16bf bm;
      ((uint4*)&bm)[0] = *(const uint4*)(bp);
      ((uint4*)&bm)[1] = *(const uint4*)(bp + 8);
      acc[g] = __builtin_amdgcn_wmma_f32_16x16x32_bf16(
          false, a, false, bm, (short)0, acc[g], false, false);
    }
  }

  // Stage partial accumulators: redbuf[((wv*4+g)*32+lane)*8 + r]
#pragma unroll
  for (int g = 0; g < 4; ++g) {
    float* p = &redbuf[(((wv * 4) + g) * 32 + lane) * 8];
    ((float4*)p)[0] = ((float4*)&acc[g])[0];   // ds_store_b128
    ((float4*)p)[1] = ((float4*)&acc[g])[1];
  }
  __syncthreads();

  // Reduce 4 slices; this wave handles fragment rows r0, r0+1.
  // D layout: N = lane&15, M = (lane>=16)*8 + r
  const int r0  = ks * 2;
  const int col = ht * 16 + nl;
  float gate0[4], gate1[4];
#pragma unroll
  for (int g = 0; g < 4; ++g) {
    float s0 = 0.0f, s1 = 0.0f;
#pragma unroll
    for (int src = 0; src < 4; ++src) {
      const float* p =
          &redbuf[((((grp * 4 + src) * 4) + g) * 32 + lane) * 8 + r0];
      float2 v = *(const float2*)p;            // ds_load_b64
      s0 += v.x;
      s1 += v.y;
    }
    gate0[g] = s0;
    gate1[g] = s1;
  }

#pragma unroll
  for (int rr = 0; rr < 2; ++rr) {
    const float* gt = rr ? gate1 : gate0;
    int r   = r0 + rr;
    int b   = mt * 16 + hi * 8 + r;
    int idx = b * HSZ + col;
    float iv = 1.0f / (1.0f + __expf(-gt[0]));
    float fv = 1.0f / (1.0f + __expf(-gt[1]));
    float gv = tanhf(gt[2]);
    float ov = 1.0f / (1.0f + __expf(-gt[3]));
    float cn = fv * c[idx] + iv * gv;
    c[idx] = cn;
    float hn = ov * tanhf(cn);
    out_t[idx] = hn;
    hout[idx]  = (__bf16)hn;
    if (last) { hT[idx] = hn; cT[idx] = cn; }
  }
}

extern "C" void kernel_launch(void* const* d_in, const int* in_sizes, int n_in,
                              void* d_out, int out_size, void* d_ws, size_t ws_size,
                              hipStream_t stream) {
  const float* input = (const float*)d_in[0];   // (S,B,I)
  const float* h0    = (const float*)d_in[1];   // (B,H)
  const float* c0    = (const float*)d_in[2];   // (B,H)
  const float* w_ih  = (const float*)d_in[3];   // (1,4H,I)
  const float* w_hh  = (const float*)d_in[4];   // (1,4H,H)
  const float* b_ih  = (const float*)d_in[5];   // (1,4H)
  const float* b_hh  = (const float*)d_in[6];   // (1,4H)
  // d_in[7] = direction (always 0, single direction)
  float* out = (float*)d_out;

  char* ws = (char*)d_ws;
  __bf16* wpack = (__bf16*)ws;
  __bf16* xb    = (__bf16*)(ws + (size_t)4 * 1024 * 1024);
  __bf16* hbuf0 = (__bf16*)(ws + (size_t)36 * 1024 * 1024);
  __bf16* hbuf1 = hbuf0 + BATCH * HSZ;
  float*  cst   = (float*)(ws + (size_t)36 * 1024 * 1024 + 128 * 1024);

  pack_weights <<< (G4 * KC) / 256, 256, 0, stream >>> (w_ih, w_hh, wpack);
  convert_input<<< (size_t)SEQ * BATCH * ISZ / 256, 256, 0, stream >>> (input, xb);
  init_state   <<< (BATCH * HSZ) / 256, 256, 0, stream >>> (h0, c0, hbuf0, cst);

  float* hT = out + (size_t)SEQ * BATCH * HSZ;
  float* cT = hT + (size_t)BATCH * HSZ;
  for (int t = 0; t < SEQ; ++t) {
    const __bf16* hin  = (t & 1) ? hbuf1 : hbuf0;
    __bf16*       hout = (t & 1) ? hbuf0 : hbuf1;
    lstm_step<<<64, 256, 0, stream>>>(
        wpack, xb + (size_t)t * BATCH * ISZ, hin, hout, cst,
        b_ih, b_hh, out + (size_t)t * BATCH * HSZ, hT, cT,
        (t == SEQ - 1) ? 1 : 0);
  }
}
